// CausalSelfAttention_40733469835724
// MI455X (gfx1250) — compile-verified, matrix operand order checked by reading, not run on hardware
//
#include <hip/hip_runtime.h>
#include <hip/hip_bf16.h>
#include <stdint.h>

typedef __bf16 bf16;
typedef __attribute__((ext_vector_type(16))) __bf16 v16bf;
typedef __attribute__((ext_vector_type(8)))  __bf16 v8bf;
typedef __attribute__((ext_vector_type(8)))  float  v8f;

#define WMMA_BF16(a,b,c) __builtin_amdgcn_wmma_f32_16x16x32_bf16(false,(a),false,(b),(short)0,(c),false,false)

static __device__ __forceinline__ v16bf cat8(v8bf lo, v8bf hi){
  return __builtin_shufflevector(lo,hi,0,1,2,3,4,5,6,7,8,9,10,11,12,13,14,15);
}
static __device__ __forceinline__ v8bf ld8(const bf16* p){
  return *reinterpret_cast<const v8bf*>(p);
}

constexpr int Bc = 2, Tc = 2048, Ec = 2048, Hc = 16, KVc = 4, Dc = 128;
constexpr int BT = Bc * Tc;
constexpr int KVE = KVc * Dc;   // 512

// ---------------- fp32 -> bf16 conversion ----------------
__global__ void cvt_f32_bf16_k(const float* __restrict__ src, bf16* __restrict__ dst, int n){
  int i = blockIdx.x * blockDim.x + threadIdx.x;
  int stride = gridDim.x * blockDim.x;
  for (; i < n; i += stride) dst[i] = (bf16)src[i];
}

// ---------------- fp32 [K][N] -> bf16 transposed [N][K] (32x32 LDS tile) ----------------
__global__ __launch_bounds__(256) void cvt_transpose_k(
    const float* __restrict__ in, bf16* __restrict__ out, int K, int N){
  __shared__ float tile[32][33];
  int n0 = blockIdx.x * 32, k0 = blockIdx.y * 32;
  int tx = threadIdx.x & 31, ty = threadIdx.x >> 5;   // 32 x 8
#pragma unroll
  for (int i = 0; i < 4; ++i)
    tile[ty + i * 8][tx] = in[(size_t)(k0 + ty + i * 8) * N + n0 + tx];
  __syncthreads();
#pragma unroll
  for (int i = 0; i < 4; ++i)
    out[(size_t)(n0 + ty + i * 8) * K + k0 + tx] = (bf16)tile[tx][ty + i * 8];
}

// ---------------- bf16 WMMA GEMM: C[MxN] f32 = A[MxK] * Bt[NxK]^T ----------------
// block = 128 threads (4 waves). Wave owns 32 rows x 64 cols (2 M-tiles, 4 N-tiles).
// All fragment loads are contiguous global_load_b128 — no LDS, no barriers.
__global__ __launch_bounds__(128) void gemm_bf16_wmma_k(
    const bf16* __restrict__ A, const bf16* __restrict__ Bt, float* __restrict__ C,
    int M, int N, int K){
  const int wave = threadIdx.x >> 5, lane = threadIdx.x & 31;
  const int half = lane >> 4, l16 = lane & 15;
  const int m0 = blockIdx.y * 128 + wave * 32;
  const int n0 = blockIdx.x * 64;

  v8f z = {0.f,0.f,0.f,0.f,0.f,0.f,0.f,0.f};
  v8f acc[2][4];
#pragma unroll
  for (int mt = 0; mt < 2; ++mt)
#pragma unroll
    for (int nt = 0; nt < 4; ++nt) acc[mt][nt] = z;

  const bf16* arow0 = A + (size_t)(m0 + l16) * K + half * 8;
  const bf16* arow1 = arow0 + (size_t)16 * K;
  const bf16* brow0 = Bt + (size_t)(n0 + l16) * K + half * 8;

  for (int k0 = 0; k0 < K; k0 += 32){
    v16bf a0 = cat8(ld8(arow0 + k0), ld8(arow0 + k0 + 16));
    v16bf a1 = cat8(ld8(arow1 + k0), ld8(arow1 + k0 + 16));
#pragma unroll
    for (int nt = 0; nt < 4; ++nt){
      const bf16* br = brow0 + (size_t)(nt * 16) * K + k0;
      v16bf bfr = cat8(ld8(br), ld8(br + 16));
      acc[0][nt] = WMMA_BF16(a0, bfr, acc[0][nt]);
      acc[1][nt] = WMMA_BF16(a1, bfr, acc[1][nt]);
    }
  }

#pragma unroll
  for (int mt = 0; mt < 2; ++mt)
#pragma unroll
    for (int nt = 0; nt < 4; ++nt)
#pragma unroll
      for (int r = 0; r < 8; ++r)
        C[(size_t)(m0 + mt * 16 + r + half * 8) * N + n0 + nt * 16 + l16] = acc[mt][nt][r];
}

// ---------------- gate = 2*sigmoid(x[:, :32] @ Wg); v = v + gate*ve -> bf16 ----------------
__global__ __launch_bounds__(512) void gate_vmix_k(
    const float* __restrict__ x, const float* __restrict__ ve,
    const float* __restrict__ Wg, const float* __restrict__ Vf,
    bf16* __restrict__ vb){
  int bt = blockIdx.x;
  int tid = threadIdx.x;          // 512 = KVc*Dc
  int c = tid >> 7;               // kv head
  const float* xr = x + (size_t)bt * Ec;
  float g = 0.f;
#pragma unroll
  for (int i = 0; i < 32; ++i) g += xr[i] * Wg[i * KVc + c];
  g = 2.f / (1.f + __expf(-g));
  size_t idx = (size_t)bt * KVE + tid;
  vb[idx] = (bf16)(Vf[idx] + g * ve[idx]);
}

// ---------------- RoPE + RMSNorm, fp32 in -> bf16 out. One wave per (token, head). ----------------
__global__ __launch_bounds__(256) void rope_rmsnorm_k(
    const float* __restrict__ in, const float* __restrict__ cosb,
    const float* __restrict__ sinb, bf16* __restrict__ out, int nheads){
  int wv   = blockIdx.x * (blockDim.x >> 5) + (threadIdx.x >> 5);
  int lane = threadIdx.x & 31;
  int total = BT * nheads;
  if (wv >= total) return;
  int bt = wv / nheads, h = wv % nheads;
  int t = bt % Tc;
  const float* row = in + ((size_t)bt * nheads + h) * Dc;
  float o[4]; float ss = 0.f;
#pragma unroll
  for (int p = 0; p < 2; ++p){
    int d = lane + p * 32;
    float t1 = row[d], t2 = row[d + 64];
    float cv = cosb[(size_t)t * 64 + d], sv = sinb[(size_t)t * 64 + d];
    float o1 =  t1 * cv + t2 * sv;
    float o2 = -t1 * sv + t2 * cv;
    o[p * 2] = o1; o[p * 2 + 1] = o2;
    ss += o1 * o1 + o2 * o2;    // rotation preserves norm
  }
#pragma unroll
  for (int off = 16; off >= 1; off >>= 1) ss += __shfl_xor(ss, off, 32);
  float inv = rsqrtf(ss * (1.f / 128.f) + 1e-6f);
  bf16* orow = out + ((size_t)bt * nheads + h) * Dc;
#pragma unroll
  for (int p = 0; p < 2; ++p){
    int d = lane + p * 32;
    orow[d]      = (bf16)(o[p * 2]     * inv);
    orow[d + 64] = (bf16)(o[p * 2 + 1] * inv);
  }
}

// ---------------- flash attention with WMMA (sliding causal window, GQA) ----------------
// grid = (T/64, B*H), block = 128. Wave w owns queries [qblk + 16w, +16).
__global__ __launch_bounds__(128) void flash_wmma_k(
    const bf16* __restrict__ Q, const bf16* __restrict__ Kb,
    const bf16* __restrict__ V, const int* __restrict__ winp,
    bf16* __restrict__ Y){
  // V block transposed: Vt[d][k], row stride 40 elems (80B, 16B-aligned)
  __shared__ __align__(16) bf16 Vt[128][40];
  // per-wave P tile [m][k], same padded stride
  __shared__ __align__(16) bf16 Ps[4][16][40];

  const int tid = threadIdx.x, wave = tid >> 5, lane = tid & 31;
  const int half = lane >> 4, l16 = lane & 15;
  const int bh = blockIdx.y, b = bh / Hc, h = bh % Hc, kvh = h / (Hc / KVc);
  const int qblk = blockIdx.x * 64;
  const int q0 = qblk + wave * 16;
  const int ws = *winp;
  const float scale = 0.08838834764831845f;  // 1/sqrt(128)

  // Q fragments held in registers for the whole kernel
  v16bf qf[4];
  {
    const bf16* qrow = Q + (((size_t)b * Tc + q0 + l16) * Hc + h) * Dc;
#pragma unroll
    for (int dc = 0; dc < 4; ++dc){
      int kbase = dc * 32 + half * 8;
      qf[dc] = cat8(ld8(qrow + kbase), ld8(qrow + kbase + 16));
    }
  }

  v8f z = {0.f,0.f,0.f,0.f,0.f,0.f,0.f,0.f};
  v8f acc[8];
#pragma unroll
  for (int i = 0; i < 8; ++i) acc[i] = z;
  float rm[8], rl[8];
#pragma unroll
  for (int r = 0; r < 8; ++r){ rm[r] = -1e30f; rl[r] = 0.f; }

  int klo = qblk - ws; if (klo < 0) klo = 0; klo &= ~31;  // uniform per workgroup
  int khi = qblk + 63;

  for (int kb = klo; kb <= khi; kb += 32){
    // stage V block transposed: 512 b128 chunks -> scattered b16 stores
#pragma unroll
    for (int it = 0; it < 4; ++it){
      int idx = tid + it * 128;
      int row = idx >> 4, ch = idx & 15;        // key row, dim chunk
      v8bf vv = ld8(V + (((size_t)b * Tc + kb + row) * KVc + kvh) * Dc + ch * 8);
#pragma unroll
      for (int i = 0; i < 8; ++i) Vt[ch * 8 + i][row] = vv[i];
    }
    __syncthreads();

    // S = Q K^T for two 16-key tiles, then online softmax
    float sst[2][8];
#pragma unroll
    for (int nt = 0; nt < 2; ++nt){
      const bf16* krow = Kb + (((size_t)b * Tc + kb + nt * 16 + l16) * KVc + kvh) * Dc;
      v8f s = z;
#pragma unroll
      for (int dc = 0; dc < 4; ++dc){
        int kbase = dc * 32 + half * 8;
        v16bf kf = cat8(ld8(krow + kbase), ld8(krow + kbase + 16));
        s = WMMA_BF16(qf[dc], kf, s);
      }
      int j = kb + nt * 16 + l16;
#pragma unroll
      for (int r = 0; r < 8; ++r){
        int i = q0 + r + half * 8;
        float sv = s[r] * scale;
        if (j > i || (i - j) > ws) sv = -1e30f;
        sst[nt][r] = sv;
      }
    }

#pragma unroll
    for (int r = 0; r < 8; ++r){
      float mx = fmaxf(sst[0][r], sst[1][r]);
#pragma unroll
      for (int off = 8; off >= 1; off >>= 1) mx = fmaxf(mx, __shfl_xor(mx, off, 16));
      float mnew = fmaxf(rm[r], mx);
      float alpha = __expf(rm[r] - mnew);
      float p0 = __expf(sst[0][r] - mnew);
      float p1 = __expf(sst[1][r] - mnew);
      float sum = p0 + p1;
#pragma unroll
      for (int off = 8; off >= 1; off >>= 1) sum += __shfl_xor(sum, off, 16);
      rl[r] = rl[r] * alpha + sum;
      rm[r] = mnew;
#pragma unroll
      for (int dch = 0; dch < 8; ++dch) acc[dch][r] *= alpha;
      // stash probabilities in A-operand-friendly [m][k] layout
      Ps[wave][r + half * 8][l16]      = (bf16)p0;
      Ps[wave][r + half * 8][16 + l16] = (bf16)p1;
    }

    // reload P as A fragment: two aligned ds_load_b128
    v16bf pa;
    {
      const bf16* pr = &Ps[wave][l16][half * 8];
      pa = cat8(ld8(pr), ld8(pr + 16));
    }

    // O += P @ V : V fragment = two aligned ds_load_b128 from transposed Vt
#pragma unroll
    for (int dch = 0; dch < 8; ++dch){
      const bf16* vr = &Vt[dch * 16 + l16][half * 8];
      v16bf vf = cat8(ld8(vr), ld8(vr + 16));
      acc[dch] = WMMA_BF16(pa, vf, acc[dch]);
    }
    __syncthreads();
  }

  // normalize and emit bf16 y in (b,t,h,d) layout for the Wo GEMM
#pragma unroll
  for (int r = 0; r < 8; ++r){
    float inv = 1.f / rl[r];
    int m = q0 + r + half * 8;
    bf16* yr = Y + (((size_t)b * Tc + m) * Hc + h) * Dc;
#pragma unroll
    for (int dch = 0; dch < 8; ++dch)
      yr[dch * 16 + l16] = (bf16)(acc[dch][r] * inv);
  }
}

// ---------------- launcher ----------------
extern "C" void kernel_launch(void* const* d_in, const int* in_sizes, int n_in,
                              void* d_out, int out_size, void* d_ws, size_t ws_size,
                              hipStream_t stream){
  (void)in_sizes; (void)n_in; (void)out_size; (void)ws_size;
  const float* x    = (const float*)d_in[0];
  const float* ve   = (const float*)d_in[1];
  const float* cosb = (const float*)d_in[2];
  const float* sinb = (const float*)d_in[3];
  const float* Wq   = (const float*)d_in[4];
  const float* Wk   = (const float*)d_in[5];
  const float* Wv   = (const float*)d_in[6];
  const float* Wo   = (const float*)d_in[7];
  const float* Wg   = (const float*)d_in[8];
  const int*   win  = (const int*)d_in[9];

  char* ws = (char*)d_ws;
  size_t off = 0;
  auto alloc = [&](size_t bytes)->char*{
    char* p = ws + off;
    off += (bytes + 255) & ~(size_t)255;
    return p;
  };
  bf16*  xb   = (bf16*) alloc((size_t)BT * Ec * 2);
  bf16*  wqt  = (bf16*) alloc((size_t)Ec * Ec * 2);    // [N][K] transposed
  bf16*  wkt  = (bf16*) alloc((size_t)Ec * KVE * 2);
  bf16*  wvt  = (bf16*) alloc((size_t)Ec * KVE * 2);
  bf16*  wot  = (bf16*) alloc((size_t)Ec * Ec * 2);
  float* Qf   = (float*)alloc((size_t)BT * Ec * 4);
  float* Kf   = (float*)alloc((size_t)BT * KVE * 4);
  float* Vf   = (float*)alloc((size_t)BT * KVE * 4);
  bf16*  qb   = (bf16*) alloc((size_t)BT * Ec * 2);
  bf16*  kb16 = (bf16*) alloc((size_t)BT * KVE * 2);
  bf16*  vb16 = (bf16*) alloc((size_t)BT * KVE * 2);
  bf16*  yb   = (bf16*)Qf;   // reuse: Qf dead after rope_rmsnorm

  // 1) convert x; transpose-convert weights to bf16 [N][K]
  cvt_f32_bf16_k<<<2048, 256, 0, stream>>>(x, xb, BT * Ec);
  cvt_transpose_k<<<dim3(Ec / 32,  Ec / 32), 256, 0, stream>>>(Wq, wqt, Ec, Ec);
  cvt_transpose_k<<<dim3(KVE / 32, Ec / 32), 256, 0, stream>>>(Wk, wkt, Ec, KVE);
  cvt_transpose_k<<<dim3(KVE / 32, Ec / 32), 256, 0, stream>>>(Wv, wvt, Ec, KVE);
  cvt_transpose_k<<<dim3(Ec / 32,  Ec / 32), 256, 0, stream>>>(Wo, wot, Ec, Ec);

  // 2) projections
  gemm_bf16_wmma_k<<<dim3(Ec / 64,  BT / 128), 128, 0, stream>>>(xb, wqt, Qf, BT, Ec,  Ec);
  gemm_bf16_wmma_k<<<dim3(KVE / 64, BT / 128), 128, 0, stream>>>(xb, wkt, Kf, BT, KVE, Ec);
  gemm_bf16_wmma_k<<<dim3(KVE / 64, BT / 128), 128, 0, stream>>>(xb, wvt, Vf, BT, KVE, Ec);

  // 3) gate + value-embedding mix (bf16 out)
  gate_vmix_k<<<BT, 512, 0, stream>>>(x, ve, Wg, Vf, vb16);

  // 4) RoPE + RMSNorm (bf16 out)
  rope_rmsnorm_k<<<(BT * Hc) / 8,  256, 0, stream>>>(Qf, cosb, sinb, qb,   Hc);
  rope_rmsnorm_k<<<(BT * KVc) / 8, 256, 0, stream>>>(Kf, cosb, sinb, kb16, KVc);

  // 5) sliding-window GQA flash attention
  flash_wmma_k<<<dim3(Tc / 64, Bc * Hc), 128, 0, stream>>>(qb, kb16, vb16, win, yb);

  // 6) output projection -> f32 d_out
  gemm_bf16_wmma_k<<<dim3(Ec / 64, BT / 128), 128, 0, stream>>>(yb, wot, (float*)d_out, BT, Ec, Ec);
}
